// local_inference_4037269258406
// MI455X (gfx1250) — matrix operand order repaired and checked by reading, not compiled, not run on hardware
//
#include <hip/hip_runtime.h>
#include <hip/hip_bf16.h>

// ---- problem constants (from reference) ----
#define L1N 512
#define L2N 512
#define BN  32
#define DN  768
#define EPSF 1e-8f

typedef __attribute__((ext_vector_type(16))) __bf16         v16bf;
typedef __attribute__((ext_vector_type(8)))  float          v8f;
typedef __attribute__((ext_vector_type(8)))  unsigned short ushort8;

// round-to-nearest-even f32 -> bf16 bits
__device__ __forceinline__ unsigned short f2bf(float f) {
    unsigned int u = __float_as_uint(f);
    u += 0x7FFFu + ((u >> 16) & 1u);
    return (unsigned short)(u >> 16);
}

// 16x16x32 bf16 fragment: 16 bf16 values per lane = two 16B chunks
struct Frag { ushort8 lo, hi; };

// A fragment: lane group hf in {0,1}; base points at row start + hf*8
// lo = K[kb + hf*8 .. +7], hi = K[kb + 16 + hf*8 .. +7]
__device__ __forceinline__ void loadA(Frag& f, const unsigned short* base, int kb) {
    f.lo = *reinterpret_cast<const ushort8*>(base + kb);
    f.hi = *reinterpret_cast<const ushort8*>(base + kb + 16);
}
// B fragment (from N-major "BT" operand): base points at col row start + hf*8
// lo = K[kb + hf*8 .. +7], hi = K[kb + 16 + hf*8 .. +7]
__device__ __forceinline__ void loadB(Frag& f, const unsigned short* base, int kb) {
    f.lo = *reinterpret_cast<const ushort8*>(base + kb);
    f.hi = *reinterpret_cast<const ushort8*>(base + kb + 16);
}

__device__ __forceinline__ v8f wstep(const Frag& a, const Frag& b, v8f c) {
    union { ushort8 u[2]; v16bf v; } ua = {{a.lo, a.hi}};
    union { ushort8 u[2]; v16bf v; } ub = {{b.lo, b.hi}};
    return __builtin_amdgcn_wmma_f32_16x16x32_bf16(
        false, ua.v, false, ub.v, (short)0, c, false, false);
}

// ---------------------------------------------------------------------------
// Kernel 1: [L,B,D] fp32 -> bf16 batch-major [B,L,D] and transposed [B,D,L]
// ---------------------------------------------------------------------------
__global__ __launch_bounds__(256) void convert_kernel(
    const float* __restrict__ x, unsigned short* __restrict__ xb,
    unsigned short* __restrict__ xT, int L) {
    size_t idx = (size_t)blockIdx.x * blockDim.x + threadIdx.x;
    size_t total = (size_t)L * BN * DN;
    if (idx >= total) return;
    int d = (int)(idx % DN);
    int b = (int)((idx / DN) % BN);
    int l = (int)(idx / ((size_t)BN * DN));
    unsigned short v = f2bf(x[idx]);
    xb[(size_t)b * L * DN + (size_t)l * DN + d] = v;
    xT[(size_t)b * DN * L + (size_t)d * L + l] = v;
}

// ---------------------------------------------------------------------------
// Kernel 2: batched bf16 WMMA GEMM, Out[M,N] = A[M,K] * BT[N,K]^T  (fp32 out)
// Block = 128 threads = 4 waves; wave computes 32 rows x 64 cols:
//   2 M-strips (A reuse of B fragments) x 4 N-tiles, 8 accumulators.
// K-loop double-buffered in registers (two k-steps per iteration; K/32 even).
// ---------------------------------------------------------------------------
__global__ __launch_bounds__(128) void wmma_gemm_kernel(
    const unsigned short* __restrict__ A, const unsigned short* __restrict__ BT,
    float* __restrict__ Out,
    long aStride, long btStride, long outStride, int K, int ldOut) {
    const int b    = blockIdx.z;
    const unsigned short* Ab = A  + (size_t)b * aStride;
    const unsigned short* Bb = BT + (size_t)b * btStride;
    float* Ob                = Out + (size_t)b * outStride;

    const int wave = threadIdx.x >> 5;
    const int lane = threadIdx.x & 31;
    const int hf   = lane >> 4;
    const int l    = lane & 15;

    const int m0 = (blockIdx.y * 4 + wave) * 32;   // 2 strips of 16 rows
    const int n0 = blockIdx.x * 64;                // 4 tiles of 16 cols

    const unsigned short* arow[2];
    arow[0] = Ab + (size_t)(m0 + l) * K + hf * 8;
    arow[1] = arow[0] + (size_t)16 * K;
    const unsigned short* brow[4];
    brow[0] = Bb + (size_t)(n0 + l) * K + hf * 8;
    brow[1] = brow[0] + (size_t)16 * K;
    brow[2] = brow[0] + (size_t)32 * K;
    brow[3] = brow[0] + (size_t)48 * K;

    v8f acc[2][4] = {};

    Frag a0[2], a1[2], b0[4], b1[4];
#pragma unroll
    for (int mt = 0; mt < 2; ++mt) loadA(a0[mt], arow[mt], 0);
#pragma unroll
    for (int nt = 0; nt < 4; ++nt) loadB(b0[nt], brow[nt], 0);

    for (int kb = 0; kb < K; kb += 64) {
        // prefetch k-step kb+32 while computing on kb
#pragma unroll
        for (int mt = 0; mt < 2; ++mt) loadA(a1[mt], arow[mt], kb + 32);
#pragma unroll
        for (int nt = 0; nt < 4; ++nt) loadB(b1[nt], brow[nt], kb + 32);
#pragma unroll
        for (int mt = 0; mt < 2; ++mt)
#pragma unroll
            for (int nt = 0; nt < 4; ++nt)
                acc[mt][nt] = wstep(a0[mt], b0[nt], acc[mt][nt]);

        // prefetch k-step kb+64 (if any) while computing on kb+32
        if (kb + 64 < K) {
#pragma unroll
            for (int mt = 0; mt < 2; ++mt) loadA(a0[mt], arow[mt], kb + 64);
#pragma unroll
            for (int nt = 0; nt < 4; ++nt) loadB(b0[nt], brow[nt], kb + 64);
        }
#pragma unroll
        for (int mt = 0; mt < 2; ++mt)
#pragma unroll
            for (int nt = 0; nt < 4; ++nt)
                acc[mt][nt] = wstep(a1[mt], b1[nt], acc[mt][nt]);
    }

#pragma unroll
    for (int mt = 0; mt < 2; ++mt)
#pragma unroll
        for (int nt = 0; nt < 4; ++nt)
#pragma unroll
            for (int r = 0; r < 8; ++r) {
                int row = m0 + mt * 16 + hf * 8 + r;
                int col = n0 + nt * 16 + l;
                Ob[(size_t)row * ldOut + col] = acc[mt][nt][r];
            }
}

// ---------------------------------------------------------------------------
// Kernel 3: row softmax over L2 with x2 mask -> alpha bf16 [B,L1,L2]
// one wave per (b,l1) row of 512; max BEFORE mask, mask applied to exp
// ---------------------------------------------------------------------------
__global__ __launch_bounds__(256) void softmax_row_kernel(
    const float* __restrict__ S, const float* __restrict__ mask2,
    unsigned short* __restrict__ alpha) {
    int wid  = blockIdx.x * (blockDim.x >> 5) + (threadIdx.x >> 5);
    int lane = threadIdx.x & 31;
    int b  = wid / L1N;
    int l1 = wid % L1N;
    const float* row = S + ((size_t)b * L1N + l1) * L2N;

    float s[16];
    float mx = -3.402823466e38f;
#pragma unroll
    for (int j = 0; j < 16; ++j) { s[j] = row[lane + 32 * j]; mx = fmaxf(mx, s[j]); }
#pragma unroll
    for (int o = 16; o > 0; o >>= 1) mx = fmaxf(mx, __shfl_xor(mx, o, 32));

    float sum = 0.f;
#pragma unroll
    for (int j = 0; j < 16; ++j) {
        int m = lane + 32 * j;
        s[j] = __expf(s[j] - mx) * mask2[(size_t)m * BN + b];
        sum += s[j];
    }
#pragma unroll
    for (int o = 16; o > 0; o >>= 1) sum += __shfl_xor(sum, o, 32);

    float inv = 1.f / (sum + EPSF);
    unsigned short* arow = alpha + ((size_t)b * L1N + l1) * L2N;
#pragma unroll
    for (int j = 0; j < 16; ++j) arow[lane + 32 * j] = f2bf(s[j] * inv);
}

// ---------------------------------------------------------------------------
// Kernel 4: column softmax over L1 with x1 mask -> beta bf16 [B,L2,L1]
// one wave per (b,m) column of S
// ---------------------------------------------------------------------------
__global__ __launch_bounds__(256) void softmax_col_kernel(
    const float* __restrict__ S, const float* __restrict__ mask1,
    unsigned short* __restrict__ beta) {
    int wid  = blockIdx.x * (blockDim.x >> 5) + (threadIdx.x >> 5);
    int lane = threadIdx.x & 31;
    int b = wid / L2N;
    int m = wid % L2N;
    const float* col = S + (size_t)b * L1N * L2N + m;

    float s[16];
    float mx = -3.402823466e38f;
#pragma unroll
    for (int j = 0; j < 16; ++j) {
        s[j] = col[(size_t)(lane + 32 * j) * L2N];
        mx = fmaxf(mx, s[j]);
    }
#pragma unroll
    for (int o = 16; o > 0; o >>= 1) mx = fmaxf(mx, __shfl_xor(mx, o, 32));

    float sum = 0.f;
#pragma unroll
    for (int j = 0; j < 16; ++j) {
        int l = lane + 32 * j;
        s[j] = __expf(s[j] - mx) * mask1[(size_t)l * BN + b];
        sum += s[j];
    }
#pragma unroll
    for (int o = 16; o > 0; o >>= 1) sum += __shfl_xor(sum, o, 32);

    float inv = 1.f / (sum + EPSF);
    unsigned short* brow = beta + ((size_t)b * L2N + m) * L1N;
#pragma unroll
    for (int j = 0; j < 16; ++j) brow[lane + 32 * j] = f2bf(s[j] * inv);
}

// ---------------------------------------------------------------------------
extern "C" void kernel_launch(void* const* d_in, const int* in_sizes, int n_in,
                              void* d_out, int out_size, void* d_ws, size_t ws_size,
                              hipStream_t stream) {
    (void)in_sizes; (void)n_in; (void)out_size; (void)ws_size;
    const float* x1 = (const float*)d_in[0];   // [L1,B,D]
    const float* m1 = (const float*)d_in[1];   // [L1,B]
    const float* x2 = (const float*)d_in[2];   // [L2,B,D]
    const float* m2 = (const float*)d_in[3];   // [L2,B]
    float* out = (float*)d_out;                // x1_dual [L1,B,D] ++ x2_dual [L2,B,D]

    // workspace layout
    char* ws = (char*)d_ws;
    const size_t szXB = (size_t)BN * L1N * DN * sizeof(unsigned short); // L1==L2
    unsigned short* x1b = (unsigned short*)(ws);
    unsigned short* x2b = (unsigned short*)(ws + 1 * szXB);
    unsigned short* x1T = (unsigned short*)(ws + 2 * szXB);
    unsigned short* x2T = (unsigned short*)(ws + 3 * szXB);
    const size_t szS = (size_t)BN * L1N * L2N * sizeof(float);
    float* S = (float*)(ws + 4 * szXB);
    const size_t szA = (size_t)BN * L1N * L2N * sizeof(unsigned short);
    unsigned short* alpha = (unsigned short*)(ws + 4 * szXB + szS);
    unsigned short* beta  = (unsigned short*)(ws + 4 * szXB + szS + szA);

    // 1) convert + transpose to bf16
    const int total = L1N * BN * DN;
    convert_kernel<<<(total + 255) / 256, 256, 0, stream>>>(x1, x1b, x1T, L1N);
    convert_kernel<<<(total + 255) / 256, 256, 0, stream>>>(x2, x2b, x2T, L2N);

    // 2) S[b] = X1b[b] (512x768) * X2b[b]^T  -> fp32 [B,L1,L2]
    {
        dim3 grid(L2N / 64, L1N / 128, BN);
        wmma_gemm_kernel<<<grid, 128, 0, stream>>>(
            x1b, x2b, S,
            (long)L1N * DN, (long)L2N * DN, (long)L1N * L2N, DN, L2N);
    }

    // 3) softmaxes
    softmax_row_kernel<<<(BN * L1N) / 8, 256, 0, stream>>>(S, m2, alpha);
    softmax_col_kernel<<<(BN * L2N) / 8, 256, 0, stream>>>(S, m1, beta);

    // 4) x1_dual = alpha (512x512) * X2b (512x768); write [L1,B,D] interleaved
    {
        dim3 grid(DN / 64, L1N / 128, BN);
        wmma_gemm_kernel<<<grid, 128, 0, stream>>>(
            alpha, x2T, out,
            (long)L1N * L2N, (long)DN * L2N, (long)DN, L2N, BN * DN);
        // 5) x2_dual = beta (512x512) * X1b (512x768); write [L2,B,D]
        wmma_gemm_kernel<<<grid, 128, 0, stream>>>(
            beta, x1T, out + (size_t)L1N * BN * DN,
            (long)L2N * L1N, (long)DN * L1N, (long)DN, L1N, BN * DN);
    }
}
